// SemanticAwareModule_76184129896981
// MI455X (gfx1250) — matrix operand order repaired and checked by reading, not compiled, not run on hardware
//
#include <hip/hip_runtime.h>
#include <hip/hip_bf16.h>

// ---------------------------------------------------------------------------
// MI455X (gfx1250) SemanticAwareModule.
// All big GEMMs -> v_wmma_f32_16x16x32_bf16 (wave32), LDS-tiled.
// * 128-thread blocks (4 waves, 2x2), wave tile 32x32 = 4 WMMA / K-step
// * no bounds guards (all dims are multiples of 64/64/32 tiles)
// * vectorized fills (global b128 loads, b64 LDS stores)
// * fragment reads = ds_load_b128 pairs (B tile stored transposed in LDS)
// * S = vf @ tf^T GEMM uses global_load_async_to_lds_b128 + double buffering
// ---------------------------------------------------------------------------

typedef __attribute__((ext_vector_type(16))) __bf16 v16bf;
typedef __attribute__((ext_vector_type(8)))  __bf16 bf16x8;
typedef __attribute__((ext_vector_type(4)))  __bf16 bf16x4;
typedef __attribute__((ext_vector_type(8)))  float  v8f;
typedef __attribute__((ext_vector_type(4)))  float  f32x4;

#define TD    256
#define BATCH 8

#define BM    64
#define BN    64
#define KS    32
#define LDT   40   // LDS tile row stride in elems (80B: 16B-aligned, bank-skewed)

union Frag { v16bf v; bf16x8 h[2]; };

__device__ inline bf16x4 load_cvt4(const float* p) {
    f32x4 v = *(const f32x4*)p;
    bf16x4 r;
    r[0] = (__bf16)v[0]; r[1] = (__bf16)v[1];
    r[2] = (__bf16)v[2]; r[3] = (__bf16)v[3];
    return r;
}
__device__ inline bf16x4 load_cvt4(const __bf16* p) { return *(const bf16x4*)p; }

// async copy of 16 bytes global -> LDS (ASYNCcnt-tracked)
__device__ inline void async_b128(unsigned lds_addr, const void* g) {
    asm volatile("global_load_async_to_lds_b128 %0, %1, off"
                 :: "v"(lds_addr), "v"((unsigned long long)(size_t)g) : "memory");
}
__device__ inline void wait_async0() {
    asm volatile("s_wait_asynccnt 0" ::: "memory");
}

// ---------------- WMMA GEMM: D = alpha*A*B (+bias) (+res), batched ---------
template <typename TA, typename TB, bool TRB, bool OUTBF, bool ASYNC>
__global__ __launch_bounds__(128) void wmma_gemm_kernel(
    const TA* __restrict__ A, long asb,
    const TB* __restrict__ Bm, long bsb,
    const float* __restrict__ bias,
    const float* __restrict__ res, long rsb,
    void* __restrict__ out, long osb,
    int M, int N, int K, int lda, int ldb, int ldc, float alpha)
{
    __shared__ __bf16 As[(ASYNC ? 2 : 1) * BM * LDT];
    __shared__ __bf16 Bt[(ASYNC ? 2 : 1) * BN * LDT];   // stored as [n][k]

    const int b  = blockIdx.z;
    const TA* Ab = A  + (long)b * asb;
    const TB* Bb = Bm + (long)b * bsb;

    const int m0   = blockIdx.y * BM;
    const int n0   = blockIdx.x * BN;
    const int tid  = threadIdx.x;
    const int lane = tid & 31;
    const int wid  = tid >> 5;          // 4 waves: 2x2
    const int wrow = (wid & 1) * 32;
    const int wcol = (wid >> 1) * 32;
    const int h    = lane >> 4;
    const int ml   = lane & 15;

    v8f acc00 = {}, acc01 = {}, acc10 = {}, acc11 = {};
    const int nIter = K / KS;

    // -------- async tile issue (bf16 sources, row-copy layouts only) -------
    auto issue_async = [&](int k0, int buf) {
        if constexpr (ASYNC) {
            unsigned ab = (unsigned)(size_t)(As + buf * BM * LDT);
            unsigned bb = (unsigned)(size_t)(Bt + buf * BN * LDT);
            #pragma unroll
            for (int i = 0; i < 2; ++i) {               // A: 64 rows x 32k
                int ch = i * 128 + tid;
                int r = ch >> 2, c8 = (ch & 3) * 8;
                async_b128(ab + (unsigned)(r * LDT + c8) * 2,
                           Ab + (long)(m0 + r) * lda + k0 + c8);
            }
            #pragma unroll
            for (int i = 0; i < 2; ++i) {               // B^T rows: n x 32k
                int ch = i * 128 + tid;
                int r = ch >> 2, c8 = (ch & 3) * 8;
                async_b128(bb + (unsigned)(r * LDT + c8) * 2,
                           Bb + (long)(n0 + r) * ldb + k0 + c8);
            }
        }
    };
    // -------- synchronous vectorized fill ---------------------------------
    auto fill = [&](int k0) {
        #pragma unroll
        for (int i = 0; i < 4; ++i) {                   // A: 64 x 32
            int vId = i * 128 + tid;
            int r = vId >> 3, c = (vId & 7) * 4;
            *(bf16x4*)&As[r * LDT + c] =
                load_cvt4(Ab + (long)(m0 + r) * lda + k0 + c);
        }
        if constexpr (TRB) {                            // B rows contiguous in k
            #pragma unroll
            for (int i = 0; i < 4; ++i) {
                int vId = i * 128 + tid;
                int r = vId >> 3, c = (vId & 7) * 4;
                *(bf16x4*)&Bt[r * LDT + c] =
                    load_cvt4(Bb + (long)(n0 + r) * ldb + k0 + c);
            }
        } else {                                        // transpose on store
            #pragma unroll
            for (int i = 0; i < 4; ++i) {
                int vId = i * 128 + tid;
                int r = vId >> 4, c = (vId & 15) * 4;   // r = k, c = n
                bf16x4 d = load_cvt4(Bb + (long)(k0 + r) * ldb + n0 + c);
                #pragma unroll
                for (int j = 0; j < 4; ++j) Bt[(c + j) * LDT + r] = d[j];
            }
        }
        if (k0 + KS < K)   // global_prefetch of next A tile rows
            __builtin_prefetch(Ab + (long)(m0 + (tid & 63)) * lda + k0 + KS, 0, 0);
    };
    // -------- fragments + 4 WMMAs -----------------------------------------
    auto compute = [&](const __bf16* as_, const __bf16* bt_) {
        Frag a0, a1, b0, b1;
        // A 16x32 frag: two contiguous 8-elem K-runs (k = 8h.. / 16+8h..)
        a0.h[0] = *(const bf16x8*)&as_[(wrow      + ml) * LDT + 8 * h];
        a0.h[1] = *(const bf16x8*)&as_[(wrow      + ml) * LDT + 16 + 8 * h];
        a1.h[0] = *(const bf16x8*)&as_[(wrow + 16 + ml) * LDT + 8 * h];
        a1.h[1] = *(const bf16x8*)&as_[(wrow + 16 + ml) * LDT + 16 + 8 * h];
        // B 32x16 frag: lane = n, two 8-elem K-runs (k = 8h.. / 16+8h..); Bt[n][k] contiguous in k
        b0.h[0] = *(const bf16x8*)&bt_[(wcol      + ml) * LDT + 8 * h];
        b0.h[1] = *(const bf16x8*)&bt_[(wcol      + ml) * LDT + 16 + 8 * h];
        b1.h[0] = *(const bf16x8*)&bt_[(wcol + 16 + ml) * LDT + 8 * h];
        b1.h[1] = *(const bf16x8*)&bt_[(wcol + 16 + ml) * LDT + 16 + 8 * h];
        acc00 = __builtin_amdgcn_wmma_f32_16x16x32_bf16(
                    false, a0.v, false, b0.v, (short)0, acc00, false, false);
        acc01 = __builtin_amdgcn_wmma_f32_16x16x32_bf16(
                    false, a0.v, false, b1.v, (short)0, acc01, false, false);
        acc10 = __builtin_amdgcn_wmma_f32_16x16x32_bf16(
                    false, a1.v, false, b0.v, (short)0, acc10, false, false);
        acc11 = __builtin_amdgcn_wmma_f32_16x16x32_bf16(
                    false, a1.v, false, b1.v, (short)0, acc11, false, false);
    };

    if constexpr (ASYNC) {
        issue_async(0, 0);
        for (int it = 0; it < nIter; ++it) {
            int cb = it & 1;
            wait_async0();                 // own ASYNCcnt drained
            __syncthreads();               // all waves' LDS writes visible
            if (it + 1 < nIter) issue_async((it + 1) * KS, cb ^ 1);
            compute(As + cb * BM * LDT, Bt + cb * BN * LDT);
            __syncthreads();
        }
    } else {
        for (int it = 0; it < nIter; ++it) {
            fill(it * KS);
            __syncthreads();
            compute(As, Bt);
            __syncthreads();
        }
    }

    // -------- epilogue: C/D layout: VGPR r, lane l: M=r+8*(l>>4), N=l&15 ---
    const float* resb = res ? res + (long)b * rsb : nullptr;
    #pragma unroll
    for (int mi = 0; mi < 2; ++mi) {
        #pragma unroll
        for (int r = 0; r < 8; ++r) {
            int gm = m0 + wrow + mi * 16 + r + 8 * h;
            float bv = bias ? bias[gm] : 0.f;
            long rowo = (long)b * osb + (long)gm * ldc;
            #pragma unroll
            for (int ni = 0; ni < 2; ++ni) {
                int gn = n0 + wcol + ni * 16 + ml;
                const v8f& a = mi ? (ni ? acc11 : acc10) : (ni ? acc01 : acc00);
                float v = a[r] * alpha + bv;
                if (resb) v += resb[(long)gm * ldc + gn];
                if (OUTBF) ((__bf16*)out)[rowo + gn] = (__bf16)v;
                else       ((float*)out)[rowo + gn] = v;
            }
        }
    }
}

// ---------------- bilinear x2 upsample (half-pixel), f32 -> bf16 -----------
__global__ __launch_bounds__(256) void upsample2x_bf16_kernel(
    const float* __restrict__ t, __bf16* __restrict__ out, int h, int total)
{
    int idx = blockIdx.x * 256 + threadIdx.x;
    if (idx >= total) return;
    int W = 2 * h;
    int x = idx % W;
    int y = (idx / W) % W;
    long bc = (long)idx / ((long)W * W);

    float sx = 0.5f * x - 0.25f;
    float sy = 0.5f * y - 0.25f;
    int x0 = (int)floorf(sx); float fx = sx - (float)x0;
    int y0 = (int)floorf(sy); float fy = sy - (float)y0;
    int x0c = x0 < 0 ? 0 : x0;  int x1c = (x0 + 1 > h - 1) ? h - 1 : x0 + 1;
    int y0c = y0 < 0 ? 0 : y0;  int y1c = (y0 + 1 > h - 1) ? h - 1 : y0 + 1;

    const float* p = t + bc * (long)h * h;
    float v00 = p[y0c * h + x0c], v01 = p[y0c * h + x1c];
    float v10 = p[y1c * h + x0c], v11 = p[y1c * h + x1c];
    float v = (1.f - fy) * ((1.f - fx) * v00 + fx * v01)
            +         fy * ((1.f - fx) * v10 + fx * v11);
    out[idx] = (__bf16)v;
}

// ---------------- row softmax over 256-wide rows, f32 -> bf16 --------------
__global__ __launch_bounds__(256) void softmax256_kernel(
    const float* __restrict__ S, __bf16* __restrict__ Sb)
{
    __shared__ float red[256];
    long row = blockIdx.x;
    int t = threadIdx.x;
    float v = S[row * 256 + t];
    red[t] = v; __syncthreads();
    for (int s = 128; s > 0; s >>= 1) {
        if (t < s) red[t] = fmaxf(red[t], red[t + s]);
        __syncthreads();
    }
    float m = red[0]; __syncthreads();
    float e = __expf(v - m);
    red[t] = e; __syncthreads();
    for (int s = 128; s > 0; s >>= 1) {
        if (t < s) red[t] += red[t + s];
        __syncthreads();
    }
    Sb[row * 256 + t] = (__bf16)(e / red[0]);
}

// ---------------- adaptive pool to 8x8 + channel concat -------------------
__global__ __launch_bounds__(256) void pool_concat_kernel(
    const float* __restrict__ t0, const float* __restrict__ t1,
    const float* __restrict__ t2, const float* __restrict__ t3,
    float* __restrict__ cat)
{
    int idx = blockIdx.x * 256 + threadIdx.x;       // [B,1024,8,8]
    if (idx >= BATCH * 1024 * 64) return;
    int x = idx & 7, y = (idx >> 3) & 7, c = (idx >> 6) & 1023, b = idx >> 16;
    int scale = c >> 8, cc = c & 255;
    const float* src; int hs;
    switch (scale) {
        case 0:  src = t0; hs = 64; break;
        case 1:  src = t1; hs = 32; break;
        case 2:  src = t2; hs = 16; break;
        default: src = t3; hs = 8;  break;
    }
    int f = hs >> 3;
    const float* p = src + ((long)b * TD + cc) * hs * hs;
    float s = 0.f;
    for (int dy = 0; dy < f; ++dy)
        for (int dx = 0; dx < f; ++dx)
            s += p[(y * f + dy) * hs + (x * f + dx)];
    cat[idx] = s / (float)(f * f);
}

// ---------------- BN batch stats (per channel over B*64 samples) ----------
__global__ __launch_bounds__(256) void bn_stats_kernel(
    const float* __restrict__ x, float* __restrict__ mu, float* __restrict__ rstd)
{
    __shared__ float s1[256], s2[256];
    int c = blockIdx.x, t = threadIdx.x;
    float a = 0.f, q = 0.f;
    for (int i = t; i < BATCH * 64; i += 256) {
        int b = i >> 6, sp = i & 63;
        float v = x[((long)b * TD + c) * 64 + sp];
        a += v; q += v * v;
    }
    s1[t] = a; s2[t] = q; __syncthreads();
    for (int s = 128; s > 0; s >>= 1) {
        if (t < s) { s1[t] += s1[t + s]; s2[t] += s2[t + s]; }
        __syncthreads();
    }
    if (t == 0) {
        float m = s1[0] / (float)(BATCH * 64);
        float var = s2[0] / (float)(BATCH * 64) - m * m;
        mu[c] = m; rstd[c] = rsqrtf(var + 1e-5f);
    }
}

// ---------------- BN apply + ReLU + spatial mean -> pooled [B,TD] ---------
__global__ __launch_bounds__(256) void bn_relu_pool_kernel(
    const float* __restrict__ x, const float* __restrict__ mu,
    const float* __restrict__ rstd, const float* __restrict__ g,
    const float* __restrict__ bt, float* __restrict__ pooled)
{
    int idx = blockIdx.x * 256 + threadIdx.x;
    if (idx >= BATCH * TD) return;
    int c = idx & 255, b = idx >> 8;
    float m = mu[c], r = rstd[c], gg = g[c], bb = bt[c];
    const float* p = x + ((long)b * TD + c) * 64;
    float s = 0.f;
    for (int i = 0; i < 64; ++i) {
        float v = (p[i] - m) * r * gg + bb;
        s += fmaxf(v, 0.f);
    }
    pooled[idx] = s / 64.f;
}

// ---------------- tiny fully-connected (M = 8, VALU path) -----------------
__global__ __launch_bounds__(256) void fc_kernel(
    const float* __restrict__ in, const float* __restrict__ W,
    const float* __restrict__ bias, float* __restrict__ out,
    int In, int Out, int relu)
{
    int idx = blockIdx.x * 256 + threadIdx.x;
    if (idx >= BATCH * Out) return;
    int o = idx % Out, b = idx / Out;
    const float* w  = W  + (long)o * In;
    const float* xi = in + (long)b * In;
    float s = bias[o];
    for (int k = 0; k < In; ++k) s = fmaf(xi[k], w[k], s);
    if (relu) s = fmaxf(s, 0.f);
    out[(long)b * Out + o] = s;
}

// ---------------------------------------------------------------------------
extern "C" void kernel_launch(void* const* d_in, const int* in_sizes, int n_in,
                              void* d_out, int out_size, void* d_ws, size_t ws_size,
                              hipStream_t stream)
{
    (void)in_sizes; (void)n_in; (void)out_size; (void)ws_size;

    const int Cs[4] = {64, 128, 256, 512};
    const int Hs[4] = {128, 64, 32, 16};         // v spatial; t spatial = Hs/2

    const float* v_in[4]; const float* t_in[4];
    const float* vpw[4];  const float* vpb[4];
    const float* opw[4];  const float* opb[4];
    for (int i = 0; i < 4; ++i) {
        v_in[i] = (const float*)d_in[i];
        t_in[i] = (const float*)d_in[4 + i];
        vpw[i]  = (const float*)d_in[8 + 4 * i];
        vpb[i]  = (const float*)d_in[9 + 4 * i];
        opw[i]  = (const float*)d_in[10 + 4 * i];
        opb[i]  = (const float*)d_in[11 + 4 * i];
    }
    const float* sgw = (const float*)d_in[24];
    const float* sgb = (const float*)d_in[25];
    const float* bng = (const float*)d_in[26];
    const float* bnb = (const float*)d_in[27];
    const float* w1  = (const float*)d_in[28];
    const float* b1  = (const float*)d_in[29];
    const float* w2  = (const float*)d_in[30];
    const float* b2  = (const float*)d_in[31];

    // -------- workspace layout --------
    const size_t BIG = (size_t)BATCH * TD * 16384 * sizeof(__bf16);  // 64 MB
    char* w = (char*)d_ws;
    __bf16* tf   = (__bf16*)(w);
    __bf16* vf   = (__bf16*)(w + BIG);
    __bf16* enh  = (__bf16*)(w + 2 * BIG);
    float*  Smat = (float*)(w + 3 * BIG);                       // 2 MB
    __bf16* Sb   = (__bf16*)(w + 3 * BIG + 2097152);            // 1 MB
    float*  cat  = (float*)(w + 3 * BIG + 2097152 + 1048576);   // 2 MB
    float*  xbuf = (float*)((char*)cat + 2097152);              // 512 KB
    float*  mu   = (float*)((char*)xbuf + 524288);
    float*  rstd = mu + TD;
    float*  pooled = rstd + TD;            // 2048 f32
    float*  fc1b   = pooled + BATCH * TD;  // 8192 f32

    float* outF = (float*)d_out;
    size_t enhOff[4];
    enhOff[0] = (size_t)BATCH * 2 * 1024;
    enhOff[1] = enhOff[0] + (size_t)BATCH * 64  * 128 * 128;
    enhOff[2] = enhOff[1] + (size_t)BATCH * 128 * 64 * 64;
    enhOff[3] = enhOff[2] + (size_t)BATCH * 256 * 32 * 32;

    // -------- per-scale cross-attention blocks --------
    for (int i = 0; i < 4; ++i) {
        const int C = Cs[i], H = Hs[i];
        const int N = H * H;
        const int th = H / 2;

        // 1) bilinear x2 upsample t -> tf (bf16 [B,TD,N])
        {
            int total = BATCH * TD * N;
            upsample2x_bf16_kernel<<<(total + 255) / 256, 256, 0, stream>>>(
                t_in[i], tf, th, total);
        }
        // 2) vf = vpw @ v + vpb   (M=TD, N=N, K=C) -> bf16
        {
            dim3 g(N / BN, TD / BM, BATCH);
            wmma_gemm_kernel<float, float, false, true, false><<<g, 128, 0, stream>>>(
                vpw[i], 0, v_in[i], (long)C * N, vpb[i], nullptr, 0,
                vf, (long)TD * N, TD, N, C, C, N, N, 1.0f);
        }
        // 3) S = vf @ tf^T / 16   (M=TD, N=TD, K=N) -> f32; async+double-buffer
        {
            dim3 g(TD / BN, TD / BM, BATCH);
            wmma_gemm_kernel<__bf16, __bf16, true, false, true><<<g, 128, 0, stream>>>(
                vf, (long)TD * N, tf, (long)TD * N, nullptr, nullptr, 0,
                Smat, (long)TD * TD, TD, TD, N, N, N, TD, 1.0f / 16.0f);
        }
        // 4) row softmax -> bf16
        softmax256_kernel<<<BATCH * TD, 256, 0, stream>>>(Smat, Sb);
        // 5) enh = softmax(S) @ tf (M=TD, N=N, K=TD) -> bf16
        {
            dim3 g(N / BN, TD / BM, BATCH);
            wmma_gemm_kernel<__bf16, __bf16, false, true, false><<<g, 128, 0, stream>>>(
                Sb, (long)TD * TD, tf, (long)TD * N, nullptr, nullptr, 0,
                enh, (long)TD * N, TD, N, TD, TD, N, N, 1.0f);
        }
        // 6) enhanced_i = opw @ enh + opb + v  (M=C, N=N, K=TD) -> f32 d_out
        {
            dim3 g(N / BN, C / BM, BATCH);
            wmma_gemm_kernel<float, __bf16, false, false, false><<<g, 128, 0, stream>>>(
                opw[i], 0, enh, (long)TD * N, opb[i], v_in[i], (long)C * N,
                outF + enhOff[i], (long)C * N, C, N, TD, TD, N, N, 1.0f);
        }
    }

    // -------- fusion head --------
    pool_concat_kernel<<<(BATCH * 1024 * 64 + 255) / 256, 256, 0, stream>>>(
        t_in[0], t_in[1], t_in[2], t_in[3], cat);
    {   // x = sgw @ cat + sgb    (M=TD, N=64, K=1024) -> f32 [B,TD,64]
        dim3 g(1, TD / BM, BATCH);
        wmma_gemm_kernel<float, float, false, false, false><<<g, 128, 0, stream>>>(
            sgw, 0, cat, (long)1024 * 64, sgb, nullptr, 0,
            xbuf, (long)TD * 64, TD, 64, 1024, 1024, 64, 64, 1.0f);
    }
    bn_stats_kernel<<<TD, 256, 0, stream>>>(xbuf, mu, rstd);
    bn_relu_pool_kernel<<<(BATCH * TD + 255) / 256, 256, 0, stream>>>(
        xbuf, mu, rstd, bng, bnb, pooled);
    fc_kernel<<<(BATCH * 1024 + 255) / 256, 256, 0, stream>>>(
        pooled, w1, b1, fc1b, TD, 1024, 1);
    fc_kernel<<<(BATCH * 2048 + 255) / 256, 256, 0, stream>>>(
        fc1b, w2, b2, outF, 1024, 2048, 0);
}
